// CosineCMLILoss_51359218925724
// MI455X (gfx1250) — matrix operand order repaired and checked by reading, not compiled, not run on hardware
//
#include <hip/hip_runtime.h>
#include <hip/hip_bf16.h>

// ---------------------------------------------------------------------------
// CosineCMLILoss for MI455X (gfx1250, wave32, WMMA)
// sim = tf(4032x768) x imf_y(196x768)^T per y, via v_wmma_f32_16x16x32_bf16.
// B K-slices staged with GLOBAL_LOAD_ASYNC_TO_LDS_B128 (ASYNCcnt) into a
// double-buffered LDS tile; B fragments double-buffered in registers so the
// WMMA pipe is not serialized behind s_wait_dscnt 0.
// ---------------------------------------------------------------------------

typedef __attribute__((ext_vector_type(16))) __bf16 v16bf;
typedef __attribute__((ext_vector_type(8)))  float  v8f;

#define BATCH 64
#define TTXT  64        // padded text rows per batch (63 real + 1 zero pad)
#define TIMG  208       // padded image rows per batch (196 real + 12 zero pad)
#define DIM   768
#define NKSTEP 24       // 768 / 32
#define BPITCH 80       // LDS row pitch (64B data + 16B bank-spread pad)
#define NEG_MAX (-3.402823466e38f)

union Frag32B { uint4 q[2]; v16bf v; };

// async memory->LDS copy of 16B per active lane (gfx1250, ASYNCcnt-tracked)
#define ASYNC_LDS_B128(ldsaddr, voff, sbase)                         \
    asm volatile("global_load_async_to_lds_b128 %0, %1, %2"          \
                 :: "v"(ldsaddr), "v"(voff), "s"(sbase) : "memory")
#define WAIT_ASYNCCNT0() asm volatile("s_wait_asynccnt 0" ::: "memory")

__device__ __forceinline__ unsigned short f2bf_rne(float f) {
    unsigned u = __float_as_uint(f);
    unsigned r = u + 0x7FFFu + ((u >> 16) & 1u);
    return (unsigned short)(r >> 16);
}

// --------------------------- prep: mask + counts ---------------------------
__global__ void k_prep_mask(const int* __restrict__ pm,
                            float* __restrict__ maskf,
                            float* __restrict__ cnt) {
    int x = threadIdx.x;
    if (x >= BATCH) return;
    int last0 = 0;                                  // index of last zero
    for (int t = 0; t < 64; ++t)
        if (pm[x * 64 + t] == 0) last0 = t;
    int c = 0;
    for (int t = 1; t < 64; ++t) {
        int v = pm[x * 64 + t];
        if (t == last0) v = 1;                      // _mask_eos
        maskf[x * TTXT + (t - 1)] = (v != 0) ? 1.0f : 0.0f;
        if (v == 0) ++c;
    }
    maskf[x * TTXT + 63] = 1.0f;                    // zero-pad row: masked
    cnt[x] = (float)c;
}

// ------------------------- bf16 packing kernels ----------------------------
__global__ void k_cvt_text(const float* __restrict__ tf,
                           unsigned short* __restrict__ A) {
    int idx = blockIdx.x * blockDim.x + threadIdx.x;   // 64*64*768
    int k = idx % DIM;
    int r = (idx / DIM) % TTXT;
    int x = idx / (DIM * TTXT);
    float v = (r < 63) ? tf[((size_t)x * 64 + (r + 1)) * DIM + k] : 0.0f;
    A[idx] = f2bf_rne(v);
}

__global__ void k_cvt_img(const float* __restrict__ imf,
                          unsigned short* __restrict__ Bp) {
    int idx = blockIdx.x * blockDim.x + threadIdx.x;   // 64*208*768
    int k = idx % DIM;
    int r = (idx / DIM) % TIMG;
    int y = idx / (DIM * TIMG);
    float v = (r < 196) ? imf[((size_t)y * 197 + (r + 1)) * DIM + k] : 0.0f;
    Bp[idx] = f2bf_rne(v);
}

// ------------------------------- main GEMM ---------------------------------
// grid = (32 x-pairs, 64 y), block = 256 (8 wave32 waves)
// wave w: x = x0 + (w>>2), M-tile = (w&3); 13 N-tiles of 16 cols each.
__global__ __launch_bounds__(256) void k_cmli_gemm(
    const unsigned short* __restrict__ Apk,
    const unsigned short* __restrict__ Bpk,
    const float* __restrict__ maskf,
    const float* __restrict__ cnt,
    float* __restrict__ t2i,
    float* __restrict__ i2t) {

    __shared__ __align__(16) unsigned char Bs[2][TIMG * BPITCH]; // double buffer
    __shared__ float cmax_s[8][TIMG];
    __shared__ float rsum_s[8];
    __shared__ float red0[256];
    __shared__ float red1[256];

    const int tid  = threadIdx.x;
    const int w    = tid >> 5;
    const int lane = tid & 31;
    const int hlf  = lane >> 4;
    const int cloc = lane & 15;
    const int y    = blockIdx.y;
    const int x0   = blockIdx.x * 2;
    const int x    = x0 + (w >> 2);
    const int mt   = w & 3;

    // valid-row bits: bit g <=> row t = mt*16 + hlf*8 + g is unmasked
    unsigned vmask = 0;
    #pragma unroll
    for (int g = 0; g < 8; ++g) {
        int t = mt * 16 + hlf * 8 + g;
        if (maskf[x * TTXT + t] == 0.0f) vmask |= (1u << g);
    }

    v8f acc[13];
    #pragma unroll
    for (int n = 0; n < 13; ++n) acc[n] = (v8f){};

    // this thread's share of the 832 x 16B staging transfers per K-slice
    int st_row[4], st_seg[4];
    bool st_ok[4];
    #pragma unroll
    for (int j = 0; j < 4; ++j) {
        int idx = tid + j * 256;
        st_ok[j]  = (idx < TIMG * 4);
        st_row[j] = idx >> 2;
        st_seg[j] = idx & 3;
    }
    const unsigned long long bbase =
        (unsigned long long)(Bpk + (size_t)y * TIMG * DIM);

    // stage one 32-K slice of B into Bs[buf] via async loads
    auto stage = [&](int buf, int kk) {
        unsigned long long sb = bbase + (unsigned)(kk * 2);
        #pragma unroll
        for (int j = 0; j < 4; ++j) {
            if (st_ok[j]) {
                unsigned goff = (unsigned)(st_row[j] * (DIM * 2) + st_seg[j] * 16);
                unsigned laddr = (unsigned)(unsigned long long)
                    (&Bs[buf][st_row[j] * BPITCH + st_seg[j] * 16]);
                ASYNC_LDS_B128(laddr, goff, sb);
            }
        }
    };

    // A row for this lane (fixed across K): M = cloc
    const unsigned short* arow =
        Apk + ((size_t)x * TTXT + mt * 16 + cloc) * DIM;
    auto load_a = [&](Frag32B& f, int kk) {
        f.q[0] = *(const uint4*)(arow + kk + hlf * 8);        // K kk..kk+7 (+8)
        f.q[1] = *(const uint4*)(arow + kk + 16 + hlf * 8);   // K kk+16..  (+8)
    };

    // prologue: stage slice 0, load A(0)
    stage(0, 0);
    Frag32B afc, afn;
    load_a(afc, 0);
    WAIT_ASYNCCNT0();
    __syncthreads();

    for (int i = 0; i < NKSTEP; ++i) {
        const int buf = i & 1;
        const unsigned char* bslice = Bs[buf];

        // issue next slice's async staging + next A frag (hidden under WMMAs)
        if (i + 1 < NKSTEP) {
            stage(buf ^ 1, (i + 1) * 32);
            load_a(afn, (i + 1) * 32);
        }

        // compute: 13 WMMAs, register-double-buffered B fragments
        Frag32B bf0, bf1;
        {
            const unsigned char* brow = bslice + cloc * BPITCH + hlf * 16;
            bf0.q[0] = *(const uint4*)(brow);
            bf0.q[1] = *(const uint4*)(brow + 32);
        }
        #pragma unroll
        for (int n = 0; n < 13; ++n) {
            Frag32B& curf = (n & 1) ? bf1 : bf0;
            Frag32B& nxtf = (n & 1) ? bf0 : bf1;
            if (n < 12) {
                const unsigned char* brow =
                    bslice + ((n + 1) * 16 + cloc) * BPITCH + hlf * 16;
                nxtf.q[0] = *(const uint4*)(brow);
                nxtf.q[1] = *(const uint4*)(brow + 32);
            }
            acc[n] = __builtin_amdgcn_wmma_f32_16x16x32_bf16(
                false, afc.v, false, curf.v, (short)0, acc[n], false, false);
        }

        WAIT_ASYNCCNT0();     // next slice fully in LDS
        __syncthreads();      // + everyone done reading current slice
        afc = afn;
    }

    // ---- row-wise: t2i_max then masked sum over rows ----
    float rsum = 0.0f;
    #pragma unroll
    for (int g = 0; g < 8; ++g) {
        float rm = acc[0][g];
        #pragma unroll
        for (int n = 1; n < 12; ++n) rm = fmaxf(rm, acc[n][g]);
        if (cloc < 4) rm = fmaxf(rm, acc[12][g]);     // cols 192..195 only
        #pragma unroll
        for (int off = 1; off < 16; off <<= 1)
            rm = fmaxf(rm, __shfl_xor(rm, off, 32));
        if ((vmask >> g) & 1u) rsum += rm;
    }
    rsum += __shfl_xor(rsum, 16, 32);
    if (lane == 0) rsum_s[w] = rsum;

    // ---- col-wise: masked max over this wave's 16 rows, per column ----
    #pragma unroll
    for (int n = 0; n < 13; ++n) {
        float cm = NEG_MAX;
        #pragma unroll
        for (int g = 0; g < 8; ++g)
            if ((vmask >> g) & 1u) cm = fmaxf(cm, acc[n][g]);
        cm = fmaxf(cm, __shfl_xor(cm, 16, 32));
        if (hlf == 0) cmax_s[w][n * 16 + cloc] = cm;
    }
    __syncthreads();

    // combine the 4 waves of each x; mean over valid columns i<196
    float v0 = 0.0f, v1 = 0.0f;
    if (tid < 196) {
        v0 = fmaxf(fmaxf(cmax_s[0][tid], cmax_s[1][tid]),
                   fmaxf(cmax_s[2][tid], cmax_s[3][tid]));
        v1 = fmaxf(fmaxf(cmax_s[4][tid], cmax_s[5][tid]),
                   fmaxf(cmax_s[6][tid], cmax_s[7][tid]));
    }
    red0[tid] = v0;
    red1[tid] = v1;
    __syncthreads();
    for (int off = 128; off > 0; off >>= 1) {
        if (tid < off) { red0[tid] += red0[tid + off]; red1[tid] += red1[tid + off]; }
        __syncthreads();
    }
    if (tid == 0) {
        i2t[(x0 + 0) * BATCH + y] = red0[0] / 196.0f;
        i2t[(x0 + 1) * BATCH + y] = red1[0] / 196.0f;
        float d0 = fmaxf(cnt[x0 + 0], 1e-6f);
        float d1 = fmaxf(cnt[x0 + 1], 1e-6f);
        t2i[(x0 + 0) * BATCH + y] =
            (rsum_s[0] + rsum_s[1] + rsum_s[2] + rsum_s[3]) / d0;
        t2i[(x0 + 1) * BATCH + y] =
            (rsum_s[4] + rsum_s[5] + rsum_s[6] + rsum_s[7]) / d1;
    }
}

// ------------------------------ final loss ---------------------------------
__global__ void k_finalize(const float* __restrict__ t2i,
                           const float* __restrict__ i2t,
                           const int* __restrict__ target,
                           float* __restrict__ out) {
    __shared__ float red[256];
    int tid = threadIdx.x;
    float s = 0.0f;
    for (int idx = tid; idx < BATCH * BATCH; idx += 256) {
        int tg = target[idx];
        float a = i2t[idx], b = t2i[idx];
        float la = (tg == 1) ? (1.0f - a) : fmaxf(a - 0.5f, 0.0f);
        float lb = (tg == 1) ? (1.0f - b) : fmaxf(b - 0.5f, 0.0f);
        s += la + lb;
    }
    red[tid] = s;
    __syncthreads();
    for (int off = 128; off > 0; off >>= 1) {
        if (tid < off) red[tid] += red[tid + off];
        __syncthreads();
    }
    if (tid == 0) out[0] = red[0] / (2.0f * (float)(BATCH * BATCH));
}

// ------------------------------- launcher ----------------------------------
extern "C" void kernel_launch(void* const* d_in, const int* in_sizes, int n_in,
                              void* d_out, int out_size, void* d_ws, size_t ws_size,
                              hipStream_t stream) {
    const float* imf    = (const float*)d_in[0];  // (64,197,768) f32
    const float* tf     = (const float*)d_in[1];  // (64,64,768)  f32
    const int*   pm     = (const int*)d_in[2];    // (64,64) i32
    const int*   target = (const int*)d_in[3];    // (64,64) i32
    float*       out    = (float*)d_out;

    char* ws = (char*)d_ws;
    unsigned short* Apk = (unsigned short*)ws;                       // 6,291,456 B
    unsigned short* Bpk = (unsigned short*)(ws + 6291456);           // 20,447,232 B
    float* maskf = (float*)(ws + 6291456 + 20447232);                // 16,384 B
    float* cntp  = maskf + BATCH * TTXT;                             // 256 B
    float* t2i   = cntp + BATCH;                                     // 16,384 B
    float* i2t   = t2i + BATCH * BATCH;                              // 16,384 B

    k_prep_mask<<<1, 64, 0, stream>>>(pm, maskf, cntp);
    k_cvt_text<<<(BATCH * TTXT * DIM) / 256, 256, 0, stream>>>(tf, Apk);
    k_cvt_img <<<(BATCH * TIMG * DIM) / 256, 256, 0, stream>>>(imf, Bpk);

    dim3 grid(BATCH / 2, BATCH);   // (32 x-pairs, 64 y)
    k_cmli_gemm<<<grid, 256, 0, stream>>>(Apk, Bpk, maskf, cntp, t2i, i2t);

    k_finalize<<<1, 256, 0, stream>>>(t2i, i2t, target, out);
}
